// DMB_46737834115118
// MI455X (gfx1250) — matrix operand order repaired and checked by reading, not compiled, not run on hardware
//
#include <hip/hip_runtime.h>
#include <math.h>

typedef __attribute__((ext_vector_type(16))) _Float16 v16h;
typedef __attribute__((ext_vector_type(4)))  _Float16 v4h;
typedef __attribute__((ext_vector_type(8)))  float    v8f;

#define BB   4
#define CHS  128
#define HH   512
#define WW   512
#define NG   32          // patches per side
#define NP   1024        // patches per image
#define DD   768         // 3*16*16
#define DD9  2304        // 9*16*16

// ---------------- K1: spectral projection (128->3) + patchify to f16 --------
// 4 pixels per thread: float4 (b128) streaming loads of hs, 8-byte f16 stores.
__global__ void k_spectral_patch(const float* __restrict__ hs,
                                 const float* __restrict__ spec,   // (128,3)
                                 _Float16* __restrict__ hp)        // (B,NP,DD)
{
    __shared__ float s_spec[CHS * 3];
    for (int i = threadIdx.x; i < CHS * 3; i += blockDim.x) s_spec[i] = spec[i];
    __syncthreads();

    int gid = blockIdx.x * blockDim.x + threadIdx.x;   // B*H*W/4 = 262,144
    if (gid >= BB * HH * (WW / 4)) return;
    int w = (gid & (WW / 4 - 1)) * 4;
    int h = (gid >> 7) & (HH - 1);
    int b = gid >> 16;

    const float4* px =
        (const float4*)(hs + ((size_t)b * CHS * HH * WW) + (size_t)h * WW + w);
    float acc[3][4] = {};
    #pragma unroll 4
    for (int c = 0; c < CHS; ++c) {
        float4 v = px[(size_t)c * (HH * WW / 4)];       // coalesced b128 per plane
        #pragma unroll
        for (int n = 0; n < 3; ++n) {
            float s = s_spec[c * 3 + n];
            acc[n][0] = fmaf(v.x, s, acc[n][0]);
            acc[n][1] = fmaf(v.y, s, acc[n][1]);
            acc[n][2] = fmaf(v.z, s, acc[n][2]);
            acc[n][3] = fmaf(v.w, s, acc[n][3]);
        }
    }
    int n1 = h >> 4, ph = h & 15, n2 = w >> 4, pw = w & 15;   // pw % 4 == 0
    int m = n1 * NG + n2;
    _Float16* dst = hp + ((size_t)(b * NP + m) * DD) + (ph * 16 + pw);
    #pragma unroll
    for (int n = 0; n < 3; ++n) {
        v4h o = { (_Float16)acc[n][0], (_Float16)acc[n][1],
                  (_Float16)acc[n][2], (_Float16)acc[n][3] };
        *(v4h*)(dst + n * 256) = o;                      // 8-byte store
    }
}

// ---------------- K2: depthwise conv, sobel grad, diff; patchify ------------
__global__ void k_ms_prep(const float* __restrict__ ms,    // (B,3,H,W)
                          const float* __restrict__ kw,    // (3,1,3,3)
                          _Float16* __restrict__ mp,       // (B,NP,DD)
                          _Float16* __restrict__ mpT,      // (B,DD,NP)
                          float* __restrict__ hardmat)     // (B,NP,DD9)
{
    int gid = blockIdx.x * blockDim.x + threadIdx.x;
    if (gid >= BB * HH * WW) return;
    int w = gid & (WW - 1);
    int h = (gid >> 9) & (HH - 1);
    int b = gid >> 18;

    int n1 = h >> 4, ph = h & 15, n2 = w >> 4, pw = w & 15;
    int m = n1 * NG + n2;
    int off = ph * 16 + pw;
    float* hrow = hardmat + (size_t)(b * NP + m) * DD9;

    for (int c = 0; c < 3; ++c) {
        const float* base = ms + (size_t)(b * 3 + c) * HH * WW;
        float v[3][3];
        #pragma unroll
        for (int di = -1; di <= 1; ++di)
            #pragma unroll
            for (int dj = -1; dj <= 1; ++dj) {
                int hh = h + di, ww2 = w + dj;
                v[di + 1][dj + 1] =
                    (hh >= 0 && hh < HH && ww2 >= 0 && ww2 < WW)
                        ? base[hh * WW + ww2] : 0.f;
            }
        float center = v[1][1];
        float gv = v[2][1] - v[0][1];                 // cross-correlation, SAME
        float gh = v[1][2] - v[1][0];
        float grad = sqrtf(gv * gv + gh * gh + 1e-6f);
        float msd = 0.f;
        #pragma unroll
        for (int i = 0; i < 3; ++i)
            #pragma unroll
            for (int j = 0; j < 3; ++j)
                msd = fmaf(v[i][j], kw[c * 9 + i * 3 + j], msd);
        float diff = center - msd;

        int k = c * 256 + off;
        mp [(size_t)(b * NP + m) * DD + k] = (_Float16)msd;
        mpT[((size_t)b * DD + k) * NP + m] = (_Float16)msd;   // transposed B
        hrow[c * 256 + off]       = center;
        hrow[(c + 3) * 256 + off] = grad;
        hrow[(c + 6) * 256 + off] = diff;
    }
}

// ---- WMMA A-fragment (layout per CDNA5 ISA 7.12.2, wave32) -----------------
__device__ __forceinline__ v16h load_a_frag(const _Float16* __restrict__ row,
                                            int k0, int hi)
{
    v16h a;
    #pragma unroll
    for (int i = 0; i < 8; ++i) {
        int k = k0 + 2 * i + 8 * hi + ((i >= 4) ? 8 : 0);
        a[2 * i]     = row[k];
        a[2 * i + 1] = row[k + 1];
    }
    return a;
}

// ---------------- K3: sim = hp (NP x DD) * mpT -> (NP x NP) f32 -------------
// One wave computes a 16x64 strip: A-fragment reused across 4 WMMAs per K-step.
__global__ void k_sim_wmma(const _Float16* __restrict__ hp,   // (B,NP,DD)
                           const _Float16* __restrict__ mpT,  // (B,DD,NP)
                           float* __restrict__ sim)           // (B,NP,NP)
{
    int b    = blockIdx.y;
    int lane = threadIdx.x & 31;
    int job  = blockIdx.x * (blockDim.x >> 5) + (threadIdx.x >> 5); // 1024/batch
    int n0 = (job >> 4) * 16;          // 64 row tiles
    int mg = (job & 15) * 64;          // m-strip of 64
    int lane16 = lane & 15, hi = lane >> 4;

    const _Float16* Arow  = hp  + (size_t)(b * NP + n0 + lane16) * DD;
    const _Float16* Bbase = mpT + (size_t)b * DD * NP + mg;

    v8f acc[4] = {};
    for (int k0 = 0; k0 < DD; k0 += 32) {
        v16h a = load_a_frag(Arow, k0, hi);
        const _Float16* bp = Bbase + (size_t)k0 * NP + lane16;  // lane16 = N column
        #pragma unroll
        for (int t = 0; t < 4; ++t) {
            v16h bf;
            #pragma unroll
            for (int s = 0; s < 16; ++s) {
                int k = (s < 8) ? (8 * hi + s) : (16 + 8 * hi + (s - 8));
                bf[s] = bp[(size_t)k * NP + t * 16];   // k-strided gather, table order
            }
            acc[t] = __builtin_amdgcn_wmma_f32_16x16x32_f16(
                false, a, false, bf, (short)0, acc[t], false, false);
        }
    }
    float* crow = sim + (size_t)b * NP * NP;
    #pragma unroll
    for (int t = 0; t < 4; ++t)
        #pragma unroll
        for (int r = 0; r < 8; ++r)
            crow[(size_t)(n0 + r + 8 * hi) * NP + mg + t * 16 + lane16] = acc[t][r];
}

// ---------------- K4: row softmax (-> f16 P) + argmax -----------------------
__global__ void k_softmax_argmax(const float* __restrict__ sim,
                                 _Float16* __restrict__ Pm,   // (B,NP,NP)
                                 int* __restrict__ hard)      // (B,NP)
{
    int n = blockIdx.x, b = blockIdx.y, t = threadIdx.x;
    const float* row = sim + (size_t)(b * NP + n) * NP;

    __shared__ float s_val[256];
    __shared__ int   s_idx[256];

    float best = -INFINITY; int bidx = 0;
    #pragma unroll
    for (int j = 0; j < NP / 256; ++j) {
        int m = t + j * 256;
        float v = row[m];
        if (v > best || (v == best && m < bidx)) { best = v; bidx = m; }
    }
    s_val[t] = best; s_idx[t] = bidx;
    __syncthreads();
    for (int s = 128; s > 0; s >>= 1) {
        if (t < s) {
            float v = s_val[t + s]; int i = s_idx[t + s];
            if (v > s_val[t] || (v == s_val[t] && i < s_idx[t])) {
                s_val[t] = v; s_idx[t] = i;
            }
        }
        __syncthreads();
    }
    float rowmax = s_val[0];
    int   rowarg = s_idx[0];
    __syncthreads();

    float e[NP / 256], lsum = 0.f;
    #pragma unroll
    for (int j = 0; j < NP / 256; ++j) {
        float v = __expf(row[t + j * 256] - rowmax);
        e[j] = v; lsum += v;
    }
    s_val[t] = lsum;
    __syncthreads();
    for (int s = 128; s > 0; s >>= 1) {
        if (t < s) s_val[t] += s_val[t + s];
        __syncthreads();
    }
    float inv = 1.f / s_val[0];
    _Float16* prow = Pm + (size_t)(b * NP + n) * NP;
    #pragma unroll
    for (int j = 0; j < NP / 256; ++j)
        prow[t + j * 256] = (_Float16)(e[j] * inv);
    if (t == 0) hard[b * NP + n] = rowarg;
}

// ---------------- K5: soft = P (NPxNP) * mp (NPxDD); write out ch 0..2 ------
// One wave computes a 16x48 strip (3 WMMAs share the A-fragment per K-step).
__global__ void k_soft_wmma(const _Float16* __restrict__ Pm,   // (B,NP,NP)
                            const _Float16* __restrict__ mp,   // (B,NP,DD)
                            float* __restrict__ out)           // (B,12,H,W)
{
    int b    = blockIdx.y;
    int lane = threadIdx.x & 31;
    int job  = blockIdx.x * (blockDim.x >> 5) + (threadIdx.x >> 5); // 1024/batch
    int n0 = (job >> 4) * 16;          // 64 row tiles
    int dg = (job & 15) * 48;          // d-strip of 48 (16 strips x 48 = 768)
    int lane16 = lane & 15, hi = lane >> 4;

    const _Float16* Arow  = Pm + (size_t)(b * NP + n0 + lane16) * NP;
    const _Float16* Bbase = mp + (size_t)b * NP * DD + dg;

    v8f acc[3] = {};
    for (int k0 = 0; k0 < NP; k0 += 32) {
        v16h a = load_a_frag(Arow, k0, hi);
        const _Float16* bp = Bbase + (size_t)k0 * DD + lane16;
        #pragma unroll
        for (int t = 0; t < 3; ++t) {
            v16h bf;
            #pragma unroll
            for (int s = 0; s < 16; ++s) {
                int k = (s < 8) ? (8 * hi + s) : (16 + 8 * hi + (s - 8));
                bf[s] = bp[(size_t)k * DD + t * 16];
            }
            acc[t] = __builtin_amdgcn_wmma_f32_16x16x32_f16(
                false, a, false, bf, (short)0, acc[t], false, false);
        }
    }
    // un-patchify directly: d -> (c,ph,pw), n -> (n1,n2)
    #pragma unroll
    for (int t = 0; t < 3; ++t) {
        int d = dg + t * 16 + lane16;
        int c = d >> 8, rem = d & 255, ph = rem >> 4, pw = rem & 15;
        #pragma unroll
        for (int r = 0; r < 8; ++r) {
            int n  = n0 + r + 8 * hi;
            int hh = (n >> 5) * 16 + ph;
            int ww = (n & 31) * 16 + pw;
            out[(((size_t)b * 12 + c) * HH + hh) * WW + ww] = acc[t][r];
        }
    }
}

// ---------------- K6: hard gather; write out ch 3..11 -----------------------
__global__ void k_gather(const float* __restrict__ hardmat,  // (B,NP,DD9)
                         const int* __restrict__ hard,       // (B,NP)
                         float* __restrict__ out)
{
    int gid = blockIdx.x * blockDim.x + threadIdx.x;         // B*NP*DD9
    if (gid >= BB * NP * DD9) return;
    int j = gid % DD9;
    int n = (gid / DD9) & (NP - 1);
    int b = gid / (DD9 * NP);
    int src = hard[b * NP + n];
    float v = hardmat[((size_t)b * NP + src) * DD9 + j];
    int c = j >> 8, rem = j & 255, ph = rem >> 4, pw = rem & 15;
    int hh = (n >> 5) * 16 + ph;
    int ww = (n & 31) * 16 + pw;
    out[(((size_t)b * 12 + (3 + c)) * HH + hh) * WW + ww] = v;
}

extern "C" void kernel_launch(void* const* d_in, const int* in_sizes, int n_in,
                              void* d_out, int out_size, void* d_ws, size_t ws_size,
                              hipStream_t stream) {
    const float* hs   = (const float*)d_in[0];   // (4,128,512,512)
    const float* ms   = (const float*)d_in[1];   // (4,3,512,512)
    const float* spec = (const float*)d_in[2];   // (128,3)
    const float* kw   = (const float*)d_in[3];   // (3,1,3,3)
    float* out = (float*)d_out;                  // (4,12,512,512)

    char* ws = (char*)d_ws;
    size_t off = 0;
    auto alloc = [&](size_t bytes) -> void* {
        void* p = ws + off;
        off = (off + bytes + 255) & ~(size_t)255;
        return p;
    };
    _Float16* hp      = (_Float16*)alloc((size_t)BB * NP * DD  * 2);  //  6 MB
    _Float16* mp      = (_Float16*)alloc((size_t)BB * NP * DD  * 2);  //  6 MB
    _Float16* mpT     = (_Float16*)alloc((size_t)BB * DD * NP  * 2);  //  6 MB
    _Float16* Pm      = (_Float16*)alloc((size_t)BB * NP * NP  * 2);  //  8 MB
    float*    sim     = (float*)   alloc((size_t)BB * NP * NP  * 4);  // 16 MB
    float*    hardmat = (float*)   alloc((size_t)BB * NP * DD9 * 4);  // 36 MB
    int*      hard    = (int*)     alloc((size_t)BB * NP * 4);

    const int npix = BB * HH * WW;                          // 1,048,576
    k_spectral_patch<<<dim3(npix / 4 / 256), 256, 0, stream>>>(hs, spec, hp);
    k_ms_prep       <<<dim3(npix / 256),     256, 0, stream>>>(ms, kw, mp, mpT, hardmat);
    k_sim_wmma      <<<dim3(1024 / 8, BB),   256, 0, stream>>>(hp, mpT, sim);
    k_softmax_argmax<<<dim3(NP, BB),         256, 0, stream>>>(sim, Pm, hard);
    k_soft_wmma     <<<dim3(1024 / 8, BB),   256, 0, stream>>>(Pm, mp, out);
    const int ng = BB * NP * DD9;                           // 9,437,184
    k_gather        <<<dim3((ng + 255) / 256), 256, 0, stream>>>(hardmat, hard, out);
}
